// CachedHeavyRecentAttentionMasker_23536420782107
// MI455X (gfx1250) — compile-verified, matrix operand order checked by reading, not run on hardware
//
#include <hip/hip_runtime.h>
#include <hip/hip_bf16.h>
#include <stdint.h>

// Problem constants (B=1)
#define Hn 32
#define LQ 512
#define LK 4096
#define KSEL 819          // int(0.2 * 4096)
#define THRESH 1638       // min(2*KSEL, int(0.75*4096))
#define NROWS (Hn * LQ)
#define WPR 128           // 4096 bits / 32 per mask row
#define NEG_MIN (-3.402823466e38f)

typedef _Float16     v16h __attribute__((ext_vector_type(16)));
typedef float        v8f  __attribute__((ext_vector_type(8)));
typedef unsigned int v4u  __attribute__((ext_vector_type(4)));
typedef int          v4i  __attribute__((ext_vector_type(4)));
typedef int          v8i  __attribute__((ext_vector_type(8)));

#if defined(__has_builtin)
#if __has_builtin(__builtin_amdgcn_tensor_load_to_lds)
#define HAVE_TDM 1
#endif
#endif

__device__ inline unsigned keyOf(float f) {
  unsigned u = __float_as_uint(f);
  // monotonic transform: larger float -> larger unsigned key
  return (u & 0x80000000u) ? ~u : (u | 0x80000000u);
}

// ---------------------------------------------------------------------------
// Kernel 1: per query row, exact top-k (k=819) membership -> 128-word bitmask.
// One workgroup (256 threads) per row; row staged in LDS via the Tensor Data
// Mover (CDNA5 TDM), then 4-pass radix select on 8-bit digits.
// ---------------------------------------------------------------------------
__global__ __launch_bounds__(256)
void k_rowselect(const float* __restrict__ scores, uint32_t* __restrict__ masks) {
  __shared__ float    rowf[LK];
  __shared__ unsigned hist[256];
  __shared__ unsigned eq[256];
  __shared__ unsigned maskw[WPR];
  __shared__ unsigned sh_digit, sh_kneed;

  const unsigned row = blockIdx.x;          // row = h*512 + q
  const unsigned t   = threadIdx.x;

#if defined(HAVE_TDM)
  if (t < 32) {  // one wave issues the DMA; EXEC is ignored by TDM
    unsigned long long ga = (unsigned long long)(const void*)(scores + (size_t)row * LK);
    unsigned ldsoff = (unsigned)(unsigned long long)(const void*)&rowf[0];
    // D# group0: count=1, lds_addr, 57-bit global addr, type=2 (bits 127:126)
    v4u g0 = { 1u, ldsoff,
               (unsigned)(ga & 0xFFFFFFFFu),
               ((unsigned)((ga >> 32) & 0x1FFFFFFu)) | 0x80000000u };
    // D# group1: data_size=2 (4B) at bits[17:16]; tensor_dim0=4096 (bits 79:48);
    // tensor_dim1=1 (bits 111:80); tile_dim0=4096 (bits 127:112);
    // tile_dim1=tile_dim2=0; tensor_dim0_stride=4096 (bits 207:160)
    v8i g1 = { (int)0x20000, (int)(4096u << 16), (int)(1u << 16),
               (int)(4096u << 16), 0, 4096, 0, 0 };
    v4i gz = {};
#if defined(__clang_major__) && (__clang_major__ >= 23)
    v8i gz8 = {};
    __builtin_amdgcn_tensor_load_to_lds(g0, g1, gz, gz, gz8, 0);
#else
    __builtin_amdgcn_tensor_load_to_lds(g0, g1, gz, gz, 0);
#endif
#if __has_builtin(__builtin_amdgcn_s_wait_tensorcnt)
    __builtin_amdgcn_s_wait_tensorcnt(0);
#else
    asm volatile("s_wait_tensorcnt 0" ::: "memory");
#endif
  }
  __syncthreads();
#else
  for (int i = t; i < LK; i += 256) rowf[i] = scores[(size_t)row * LK + i];
  __syncthreads();
#endif

  // ---- 4-pass MSB-first radix select for the KSEL-th largest key ----
  unsigned prefix = 0, kneed = KSEL;
  for (int shift = 24; shift >= 0; shift -= 8) {
    hist[t] = 0;
    __syncthreads();
    const unsigned hiMask = (shift == 24) ? 0u : (0xFFFFFFFFu << (shift + 8));
    for (int i = 0; i < 16; ++i) {
      unsigned kk = keyOf(rowf[i * 256 + t]);
      if ((kk & hiMask) == (prefix & hiMask))
        atomicAdd(&hist[(kk >> shift) & 255u], 1u);
    }
    __syncthreads();
    if (t == 0) {
      unsigned acc = 0; int d = 255;
      for (;;) { acc += hist[d]; if (acc >= kneed || d == 0) break; --d; }
      sh_digit = (unsigned)d;
      sh_kneed = kneed - (acc - hist[d]);   // how many equals still needed
    }
    __syncthreads();
    prefix |= sh_digit << shift;
    kneed   = sh_kneed;
  }
  const unsigned T    = prefix;  // exact k-th largest key
  const unsigned kfin = kneed;   // #elements == T to include (lowest index first)

  // ---- membership with exact tie-break in index order ----
  if (t < WPR) maskw[t] = 0;
  unsigned loc = 0;
  for (int i = 0; i < 16; ++i)
    if (keyOf(rowf[t * 16 + i]) == T) ++loc;
  eq[t] = loc;
  __syncthreads();
  if (t == 0) {
    unsigned run = 0;
    for (int j = 0; j < 256; ++j) { unsigned c = eq[j]; eq[j] = run; run += c; }
  }
  __syncthreads();
  unsigned r = eq[t];
  for (int i = 0; i < 16; ++i) {
    const int e = t * 16 + i;
    const unsigned kk = keyOf(rowf[e]);
    bool m = false;
    if (kk > T)       m = true;
    else if (kk == T) { m = (r < kfin); ++r; }
    if (m) atomicOr(&maskw[e >> 5], 1u << (e & 31));
  }
  __syncthreads();
  if (t < WPR) masks[(size_t)row * WPR + t] = maskw[t];
}

// ---------------------------------------------------------------------------
// Kernel 2: per head, back-to-front union scan; first n with |union|>=1638.
// One wave per head; 4 union words per lane; shuffle reduction for popcount.
// ---------------------------------------------------------------------------
__global__ __launch_bounds__(32)
void k_headscan(const uint32_t* __restrict__ masks, int* __restrict__ nsel,
                float* __restrict__ cnt) {
  const int h    = blockIdx.x;
  const int lane = threadIdx.x;
  if (h == 0 && lane == 0) *cnt = 0.0f;   // zero density accumulator for kernel 3
  unsigned u0 = 0, u1 = 0, u2 = 0, u3 = 0;
  int found = 0;
  for (int n = 1; n <= LQ; ++n) {
    const int q = LQ - n;
    const uint32_t* rp = masks + ((size_t)(h * LQ + q)) * WPR;
    if (q > 0) __builtin_prefetch(rp - WPR, 0, 1);   // global_prefetch_b8
    const uint4 m = ((const uint4*)rp)[lane];
    u0 |= m.x; u1 |= m.y; u2 |= m.z; u3 |= m.w;
    int pc = __popc(u0) + __popc(u1) + __popc(u2) + __popc(u3);
    for (int off = 16; off > 0; off >>= 1) pc += __shfl_xor(pc, off, 32);
    if (pc >= THRESH) { found = n; break; }
  }
  if (lane == 0) nsel[h] = found;   // 0 => not chosen
}

// ---------------------------------------------------------------------------
// Kernel 3: per head-group (4 heads), union of last n_stop rows, group-OR,
// density popcount. The 32-lane popcount partials are reduced with a real
// v_wmma_f32_16x16x32_f16 (B = ones => exact integer row sums).
// ---------------------------------------------------------------------------
__global__ __launch_bounds__(128)
void k_groupunion(const uint32_t* __restrict__ masks, const int* __restrict__ nsel,
                  uint32_t* __restrict__ gun, float* __restrict__ cnt) {
  __shared__ int sh_nstop;
  const int g = blockIdx.x;
  const int t = threadIdx.x;   // word index 0..127
  if (t == 0) {
    int mx = 0; bool all = true;
    for (int j = 0; j < Hn; ++j) {
      const int v = nsel[j];
      if (v == 0) all = false; else mx = (v > mx) ? v : mx;
    }
    sh_nstop = all ? mx : LQ;
  }
  __syncthreads();
  const int nstop = sh_nstop;
  unsigned w0 = 0, w1 = 0, w2 = 0, w3 = 0;
  const size_t headStride = (size_t)LQ * WPR;
  for (int q = LQ - nstop; q < LQ; ++q) {
    const size_t base = ((size_t)(g * 4) * LQ + q) * WPR + t;
    w0 |= masks[base];
    w1 |= masks[base + headStride];
    w2 |= masks[base + 2 * headStride];
    w3 |= masks[base + 3 * headStride];
  }
  const unsigned gw = w0 | w1 | w2 | w3;
  gun[g * WPR + t] = gw;

  // density partial: sum of 32 lane popcounts via WMMA (exact: values <= 32)
  const float s = (float)__popc(gw);
  v16h a = {}; a[0] = (_Float16)s;
  v16h b;
  for (int i = 0; i < 16; ++i) b[i] = (_Float16)1.0f;
  v8f c = {};
  c = __builtin_amdgcn_wmma_f32_16x16x32_f16(false, a, false, b, (short)0, c,
                                             false, false);
  float part = c[0] + c[1] + c[2] + c[3] + c[4] + c[5] + c[6] + c[7];
  const float tot = part + __shfl_xor(part, 16, 32);   // all 32 lane values
  if ((t & 31) == 0) atomicAdd(cnt, tot);
}

// ---------------------------------------------------------------------------
// Kernel 4: write 256 MB output. All zeros except last query row per head:
// 0 where selected, -FLT_MAX otherwise. Plus the density scalar.
// ---------------------------------------------------------------------------
__global__ __launch_bounds__(256)
void k_writeout(const uint32_t* __restrict__ gun, const float* __restrict__ cnt,
                float* __restrict__ out) {
  const size_t i4 = (size_t)blockIdx.x * 256 + threadIdx.x;
  const size_t e  = i4 * 4;                 // element index, multiple of 4
  const int l = (int)(e & (size_t)(LK - 1));
  const int q = (int)((e >> 12) & (size_t)(LQ - 1));
  float4 v = make_float4(0.f, 0.f, 0.f, 0.f);
  if (q == LQ - 1) {
    const int h = (int)(e >> 21);           // e / (LK*LQ)
    const unsigned w = gun[(h >> 2) * WPR + (l >> 5)];
    const int bb = l & 31;                  // l % 4 == 0, so bb+3 <= 31
    v.x = ((w >> bb)       & 1u) ? 0.f : NEG_MIN;
    v.y = ((w >> (bb + 1)) & 1u) ? 0.f : NEG_MIN;
    v.z = ((w >> (bb + 2)) & 1u) ? 0.f : NEG_MIN;
    v.w = ((w >> (bb + 3)) & 1u) ? 0.f : NEG_MIN;
  }
  ((float4*)out)[i4] = v;
  if (i4 == 0) out[(size_t)Hn * LQ * LK] = (*cnt) * (1.0f / 32768.0f);
}

// ---------------------------------------------------------------------------
extern "C" void kernel_launch(void* const* d_in, const int* in_sizes, int n_in,
                              void* d_out, int out_size, void* d_ws, size_t ws_size,
                              hipStream_t stream) {
  (void)in_sizes; (void)n_in; (void)out_size; (void)ws_size;
  const float* scores = (const float*)d_in[0];
  char* wsb = (char*)d_ws;
  uint32_t* masks = (uint32_t*)wsb;                                  // 8 MB
  int*      nsel  = (int*)(wsb + (size_t)NROWS * WPR * 4);           // 128 B
  uint32_t* gun   = (uint32_t*)(wsb + (size_t)NROWS * WPR * 4 + 128);// 4 KB
  float*    cnt   = (float*)(wsb + (size_t)NROWS * WPR * 4 + 128 + 8 * WPR * 4);
  float*    out   = (float*)d_out;

  k_rowselect <<<NROWS, 256, 0, stream>>>(scores, masks);
  k_headscan  <<<Hn,     32, 0, stream>>>(masks, nsel, cnt);
  k_groupunion<<<Hn / 4, 128, 0, stream>>>(masks, nsel, gun, cnt);
  k_writeout  <<<(unsigned)(((size_t)Hn * LQ * LK) / (4 * 256)), 256, 0, stream>>>(gun, cnt, out);
}
